// SelfAttention_27487790694979
// MI455X (gfx1250) — matrix operand order repaired and checked by reading, not compiled, not run on hardware
//
#include <hip/hip_runtime.h>
#include <math.h>

typedef __attribute__((ext_vector_type(16))) __bf16 v16bf;
typedef __attribute__((ext_vector_type(8)))  float  v8f;
typedef __attribute__((ext_vector_type(4)))  int    i4v;

#define HDIM 256
#define LDIM 1024
#define BDIM 16
#define MTOT (BDIM * LDIM)        // 16384 rows of X
#define WPAD 8                     // bf16 pad per LDS row (bank spread)
#define WROW (HDIM + WPAD)         // 264 bf16 = 528 bytes per row
#define K1_LDS_BYTES (HDIM * WROW * 2)   // 135168 B < 320 KB/WGP

#if defined(__has_builtin)
#  if __has_builtin(__builtin_amdgcn_global_load_async_to_lds_b128)
#    define HAS_ASYNC_LDS 1
#  else
#    define HAS_ASYNC_LDS 0
#  endif
#else
#  define HAS_ASYNC_LDS 0
#endif

// Native CDNA5 transcendentals (v_tanh_f32 / v_exp_f32), guarded.
__device__ __forceinline__ float fast_tanh(float x) {
#if __has_builtin(__builtin_amdgcn_tanhf)
    return __builtin_amdgcn_tanhf(x);
#else
    float r;
    asm("v_tanh_f32 %0, %1\n\tv_nop\n\tv_nop" : "=v"(r) : "v"(x));
    return r;
#endif
}
__device__ __forceinline__ float fast_exp2(float x) {
#if __has_builtin(__builtin_amdgcn_exp2f)
    return __builtin_amdgcn_exp2f(x);
#else
    float r;
    asm("v_exp_f32 %0, %1\n\tv_nop\n\tv_nop" : "=v"(r) : "v"(x));
    return r;
#endif
}
#define LOG2E 1.4426950408889634f

__device__ __forceinline__ void fill8(v16bf& a, int base, float4 u, float4 v) {
    a[base + 0] = (__bf16)u.x; a[base + 1] = (__bf16)u.y;
    a[base + 2] = (__bf16)u.z; a[base + 3] = (__bf16)u.w;
    a[base + 4] = (__bf16)v.x; a[base + 5] = (__bf16)v.y;
    a[base + 6] = (__bf16)v.z; a[base + 7] = (__bf16)v.w;
}

// ---------------------------------------------------------------------------
// K0: Wsum = bf16(Wj + Wt), packed (256x256), written once and reused by all
// 128 K1 blocks (stays L2-resident: 128 KB).
// ---------------------------------------------------------------------------
__global__ __launch_bounds__(256) void k0_wsum(const float* __restrict__ Wj,
                                               const float* __restrict__ Wt,
                                               unsigned int* __restrict__ wsum) {
    const int i4 = blockIdx.x * 256 + threadIdx.x;   // float4 index, 0..16383
    const float4 a = ((const float4*)Wj)[i4];
    const float4 b = ((const float4*)Wt)[i4];
    union { __bf16 h[2]; unsigned int u; } p0, p1;
    p0.h[0] = (__bf16)(a.x + b.x); p0.h[1] = (__bf16)(a.y + b.y);
    p1.h[0] = (__bf16)(a.z + b.z); p1.h[1] = (__bf16)(a.w + b.w);
    uint2 o; o.x = p0.u; o.y = p1.u;
    ((uint2*)wsum)[i4] = o;
}

// ---------------------------------------------------------------------------
// K1: v[b,l] = sum_h w[h] * tanh( sum_k Wsum[h,k] * X[b,l,k] )
// 128 blocks x 8 waves; wave computes 16 rows with v_wmma_f32_16x16x32_bf16.
// Wsum staged into LDS via async global->LDS b128 copies (ASYNCcnt).
// ---------------------------------------------------------------------------
__global__ __launch_bounds__(256) void k1_v(const float* __restrict__ X,
                                            const float* __restrict__ w,
                                            const unsigned int* __restrict__ wsum,
                                            float* __restrict__ v_out) {
    extern __shared__ __bf16 Ws[];   // [HDIM][WROW] bf16, row-major, padded
    const int tid = threadIdx.x;

    // Stage Wsum (131072 B) into padded LDS rows: 8192 x b128 chunks.
#if HAS_ASYNC_LDS
    typedef __attribute__((address_space(1))) i4v GA;   // global int4
    typedef __attribute__((address_space(3))) i4v LA;   // LDS int4
    #pragma unroll 4
    for (int j = 0; j < 32; ++j) {
        const int chunk = j * 256 + tid;          // 0..8191
        const int row  = chunk >> 5;              // 32 chunks per 512B row
        const int colc = chunk & 31;
        const char* gsrc = (const char*)wsum + (size_t)chunk * 16;
        char* ldst = (char*)Ws + row * (WROW * 2) + colc * 16;
        __builtin_amdgcn_global_load_async_to_lds_b128((GA*)gsrc, (LA*)ldst, 0, 0);
    }
    asm volatile("s_wait_asynccnt 0x0" ::: "memory");
#else
    for (int j = 0; j < 32; ++j) {
        const int chunk = j * 256 + tid;
        const int row  = chunk >> 5;
        const int colc = chunk & 31;
        const uint4 d = ((const uint4*)wsum)[chunk];
        *(uint4*)((char*)Ws + row * (WROW * 2) + colc * 16) = d;
    }
#endif
    __syncthreads();

    const int lane = tid & 31;
    const int wave = tid >> 5;
    const int half = lane >> 4;            // 0: lanes 0-15, 1: lanes 16-31
    const int l16  = lane & 15;
    const int mw   = blockIdx.x * 128 + wave * 16;   // first row of this wave
    const int myRow = mw + l16;

    // Preload A fragments for all K=256 (8 steps of 32).
    // ISA 16-bit A 16x32 layout: lanes<16 hold K {0-7,16-23}, lanes>=16 {8-15,24-31}.
    v16bf aF[8];
    const float* xrow = X + (size_t)myRow * HDIM;
    #pragma unroll
    for (int ks = 0; ks < 8; ++ks) {
        const int c0 = ks * 32 + half * 8;
        const float4 f0 = *(const float4*)(xrow + c0);
        const float4 f1 = *(const float4*)(xrow + c0 + 4);
        const float4 f2 = *(const float4*)(xrow + c0 + 16);
        const float4 f3 = *(const float4*)(xrow + c0 + 20);
        fill8(aF[ks], 0, f0, f1);
        fill8(aF[ks], 8, f2, f3);
    }

    float vacc[8] = {0.f, 0.f, 0.f, 0.f, 0.f, 0.f, 0.f, 0.f};

    for (int nt = 0; nt < 16; ++nt) {           // 16 N-tiles cover H=256
        const int hcol = nt * 16 + l16;         // this lane's output column
        // B layout: lane n (=l16) holds B[k][n] = Wsum[hcol][k];
        // lanes<16 get K {0-7,16-23}, lanes>=16 get K {8-15,24-31} of each 32-step.
        const uint4* srcRow = (const uint4*)(Ws + hcol * WROW);
        union BU { uint4 q[2]; v16bf v; };
        BU bu[2];
        bu[0].q[0] = srcRow[half + 0];
        bu[0].q[1] = srcRow[half + 2];
        v8f c = {};
        #pragma unroll
        for (int ks = 0; ks < 8; ++ks) {        // K = 256 in steps of 32
            const int cb = ks & 1, nb = cb ^ 1;
            if (ks < 7) {                        // double-buffer next B frag
                bu[nb].q[0] = srcRow[(ks + 1) * 4 + half + 0];
                bu[nb].q[1] = srcRow[(ks + 1) * 4 + half + 2];
            }
            c = __builtin_amdgcn_wmma_f32_16x16x32_bf16(
                    false, aF[ks], false, bu[cb].v, (short)0, c, false, false);
        }
        // proj tile done: native v_tanh_f32 + dot with w[hcol].
        const float wcol = w[hcol];
        #pragma unroll
        for (int r = 0; r < 8; ++r)
            vacc[r] += wcol * fast_tanh(c[r]);
    }

    // Reduce across the 16 lanes sharing each row (xor masks stay in-half).
    #pragma unroll
    for (int r = 0; r < 8; ++r) {
        float s = vacc[r];
        s += __shfl_xor(s, 1, 32);
        s += __shfl_xor(s, 2, 32);
        s += __shfl_xor(s, 4, 32);
        s += __shfl_xor(s, 8, 32);
        vacc[r] = s;
    }
    // C layout: lanes<16 hold M=r (rows mw..mw+7), lanes>=16 hold M=r+8.
    if (l16 == 0) {
        #pragma unroll
        for (int r = 0; r < 8; ++r)
            v_out[mw + half * 8 + r] = vacc[r];
    }
}

// ---------------------------------------------------------------------------
// K2: per-batch softmax over v[b,:], then partial ctx over a 128-row L-chunk.
// grid (8, 16) = (L-chunk, batch), 256 threads (thread = h column).
// ---------------------------------------------------------------------------
__global__ __launch_bounds__(256) void k2_softmax_partial(
        const float* __restrict__ X, const float* __restrict__ v,
        float* __restrict__ part) {
    const int b  = blockIdx.y;
    const int lc = blockIdx.x;
    const int t  = threadIdx.x;
    __shared__ float red[256];
    __shared__ float pch[128];

    const float* vb = v + b * LDIM;
    const float4 mv = ((const float4*)vb)[t];            // 4 values per thread
    float m = fmaxf(fmaxf(mv.x, mv.y), fmaxf(mv.z, mv.w));
    red[t] = m; __syncthreads();
    for (int s = 128; s > 0; s >>= 1) {
        if (t < s) red[t] = fmaxf(red[t], red[t + s]);
        __syncthreads();
    }
    const float vmax = red[0]; __syncthreads();

    // softmax in base-2: exp(x) = exp2(x * log2e), native v_exp_f32.
    float se = fast_exp2((mv.x - vmax) * LOG2E) + fast_exp2((mv.y - vmax) * LOG2E)
             + fast_exp2((mv.z - vmax) * LOG2E) + fast_exp2((mv.w - vmax) * LOG2E);
    red[t] = se; __syncthreads();
    for (int s = 128; s > 0; s >>= 1) {
        if (t < s) red[t] += red[t + s];
        __syncthreads();
    }
    const float rinv = 1.0f / red[0];

    if (t < 128) pch[t] = fast_exp2((vb[lc * 128 + t] - vmax) * LOG2E) * rinv;
    __syncthreads();

    // partial ctx[h] over this chunk's 128 rows; coalesced across h = t.
    const float* xb = X + ((size_t)b * LDIM + lc * 128) * HDIM;
    float acc = 0.f;
    for (int l = 0; l < 128; ++l)
        acc = fmaf(pch[l], xb[(size_t)l * HDIM + t], acc);
    part[(b * 8 + lc) * HDIM + t] = acc;
}

// ---------------------------------------------------------------------------
// K3: ctx[b,h] = sum of 8 partials (L2-hot), broadcast to 64 output rows.
// 256 blocks x 256 threads; b128 coalesced stores.
// ---------------------------------------------------------------------------
__global__ __launch_bounds__(256) void k3_bcast(const float* __restrict__ part,
                                                float* __restrict__ out) {
    const int blk = blockIdx.x;          // 0..255
    const int b   = blk >> 4;
    const int l0  = (blk & 15) * 64;
    const int t   = threadIdx.x;
    __shared__ float ctx[HDIM];

    float s = 0.f;
    #pragma unroll
    for (int lc = 0; lc < 8; ++lc)
        s += part[(b * 8 + lc) * HDIM + t];
    ctx[t] = s;
    __syncthreads();

    const int h4 = t & 63;               // float4 column
    const int r0 = t >> 6;               // 4 rows per pass
    const float4 val = ((const float4*)ctx)[h4];
    float4* ob = (float4*)(out + ((size_t)(b * LDIM + l0)) * HDIM);
    #pragma unroll
    for (int rr = 0; rr < 16; ++rr)
        ob[(size_t)(r0 + rr * 4) * 64 + h4] = val;
}

extern "C" void kernel_launch(void* const* d_in, const int* in_sizes, int n_in,
                              void* d_out, int out_size, void* d_ws, size_t ws_size,
                              hipStream_t stream) {
    const float* X  = (const float*)d_in[0];   // (B,L,H) fp32
    const float* w  = (const float*)d_in[1];   // (1,H)
    const float* Wj = (const float*)d_in[2];   // (H,H)
    const float* Wt = (const float*)d_in[3];   // (H,H)
    float* out = (float*)d_out;                // (B,L,1,1,H) fp32

    float* v    = (float*)d_ws;                        // 16384 floats
    float* part = v + MTOT;                            // 32768 floats
    unsigned int* wsum = (unsigned int*)(part + 128 * HDIM);  // 128 KB bf16

    k0_wsum<<<64, 256, 0, stream>>>(Wj, Wt, wsum);
    k1_v<<<MTOT / 128, 256, K1_LDS_BYTES, stream>>>(X, w, wsum, v);
    k2_softmax_partial<<<dim3(8, BDIM), 256, 0, stream>>>(X, v, part);
    k3_bcast<<<256, 256, 0, stream>>>(part, out);
}
